// LearnedKeyQueryRetriever_12111807775121
// MI455X (gfx1250) — matrix operand order repaired and checked by reading, not compiled, not run on hardware
//
#include <hip/hip_runtime.h>

typedef __attribute__((ext_vector_type(16))) _Float16 v16h;
typedef __attribute__((ext_vector_type(8)))  _Float16 v8h;
typedef __attribute__((ext_vector_type(8)))  float    v8f;

#define NB          256       // queries
#define QD          128
#define KDIM        64
#define NCORPUS     1000000
#define TOPK        10
#define TILES_TOTAL (NCORPUS / 16)   // 62500 exact
#define QG          2         // query tiles per wave (8 waves x 32 q = 256)
#define WAVES       8
#define STAGE_TILES 16        // corpus tiles staged in LDS per step
#define ROW_STRIDE  72        // halves per row in LDS (64 + 8 pad, 144B)
#define MAX_SLICES  384

// ---------------------------------------------------------------------------
// Branchless unrolled insertion into a register-resident descending top-10.
// Caller guards with (sc > s[TOPK-1]) so this runs rarely.
// ---------------------------------------------------------------------------
__device__ __forceinline__ void ins10(float (&s)[TOPK], int (&ix)[TOPK],
                                      float sc, int id) {
    float cs = sc; int ci = id;
#pragma unroll
    for (int j = 0; j < TOPK; ++j) {
        bool gt = cs > s[j];
        float os = s[j]; int oi = ix[j];
        s[j]  = gt ? cs : os;
        ix[j] = gt ? ci : oi;
        cs    = gt ? os : cs;
        ci    = gt ? oi : ci;
    }
}

// Filter 8 WMMA accumulator scores into a per-lane top-10.
__device__ __forceinline__ void filter8(const v8f& c, int idbase,
                                        float (&s)[TOPK], int (&ix)[TOPK]) {
    float mx = c[0];
#pragma unroll
    for (int r = 1; r < 8; ++r) mx = fmaxf(mx, c[r]);
    if (mx > s[TOPK - 1]) {
#pragma unroll
        for (int r = 0; r < 8; ++r) {
            float sc = c[r];
            if (sc > s[TOPK - 1]) ins10(s, ix, sc, idbase + r);
        }
    }
}

// ---------------------------------------------------------------------------
// Kernel 1: projected[b][k] = sum_i queries[b][i] * W[k][i], stored as f16.
// ---------------------------------------------------------------------------
__global__ __launch_bounds__(256)
void proj_kernel(const float* __restrict__ q, const float* __restrict__ w,
                 _Float16* __restrict__ proj) {
    __shared__ float wl[KDIM * QD];     // 32 KB
    const int tid = threadIdx.x;
    for (int i = tid; i < KDIM * QD; i += 256) wl[i] = w[i];
    __syncthreads();

    const int b = tid;                  // one query row per thread
    float acc[KDIM];
#pragma unroll
    for (int k = 0; k < KDIM; ++k) acc[k] = 0.f;
    for (int i = 0; i < QD; ++i) {
        float qv = q[(size_t)b * QD + i];
#pragma unroll
        for (int k = 0; k < KDIM; ++k) acc[k] += qv * wl[k * QD + i];
    }
#pragma unroll
    for (int k = 0; k < KDIM; ++k) proj[(size_t)b * KDIM + k] = (_Float16)acc[k];
}

// ---------------------------------------------------------------------------
// Kernel 2: corpus streamed from HBM exactly once.
//   Stage: 256 threads convert 16 tiles (256 rows) f32->f16 into LDS.
//   Compute: wave wv serves q-tiles {2wv, 2wv+1}; A tiles read from LDS.
//   A (SRC0) = 16 corpus rows x K ; B (SRC1) = K x 16 queries (resident).
//   C lane layout: N = lane%16 (fixed query), M = r + 8*(lane/16).
// ---------------------------------------------------------------------------
__global__ __launch_bounds__(256)
void score_topk_kernel(const float* __restrict__ corpus,
                       const _Float16* __restrict__ proj,
                       float* __restrict__ cand_s, int* __restrict__ cand_i,
                       int nslices, int tps) {
    __shared__ _Float16 buf[STAGE_TILES * 16 * ROW_STRIDE];   // 36 KB

    const int tid   = threadIdx.x;
    const int lane  = tid & 31;
    const int wv    = tid >> 5;
    const int h     = lane >> 4;      // lane half: selects K sub-range
    const int m16   = lane & 15;      // A: row-in-tile, B/C: query-in-tile
    const int slice = blockIdx.x;

    // Resident B operands: lane holds proj[q][kt*32 + 16h .. +15]
    v16h bq[QG][2];
#pragma unroll
    for (int g = 0; g < QG; ++g)
#pragma unroll
        for (int kt = 0; kt < 2; ++kt)
            bq[g][kt] = *(const v16h*)(proj
                        + (size_t)((wv * QG + g) * 16 + m16) * KDIM
                        + kt * 32 + 16 * h);

    float ts[QG][TOPK]; int ti[QG][TOPK];
#pragma unroll
    for (int g = 0; g < QG; ++g)
#pragma unroll
        for (int j = 0; j < TOPK; ++j) { ts[g][j] = -3.0e38f; ti[g][j] = 0; }

    const int t0 = slice * tps;
    int t1 = t0 + tps; if (t1 > TILES_TOTAL) t1 = TILES_TOTAL;

    for (int sb = t0; sb < t1; sb += STAGE_TILES) {
        __syncthreads();   // previous stage fully consumed
        // ---- stage: thread tid handles one corpus row (rowslot == tid) ----
        {
            const int tl = tid >> 4;            // tile-local 0..15
            if (sb + tl < t1) {
                const int grow = (sb + tl) * 16 + (tid & 15);
                const float4* rp = (const float4*)(corpus + (size_t)grow * KDIM);
                _Float16* dst = &buf[tid * ROW_STRIDE];
#pragma unroll
                for (int seg = 0; seg < 4; ++seg) {     // 16 halves per seg
                    float4 fa = rp[seg * 4 + 0];
                    float4 fb = rp[seg * 4 + 1];
                    float4 fc = rp[seg * 4 + 2];
                    float4 fd = rp[seg * 4 + 3];
                    v8h p0, p1;
                    p0[0]=(_Float16)fa.x; p0[1]=(_Float16)fa.y;
                    p0[2]=(_Float16)fa.z; p0[3]=(_Float16)fa.w;
                    p0[4]=(_Float16)fb.x; p0[5]=(_Float16)fb.y;
                    p0[6]=(_Float16)fb.z; p0[7]=(_Float16)fb.w;
                    p1[0]=(_Float16)fc.x; p1[1]=(_Float16)fc.y;
                    p1[2]=(_Float16)fc.z; p1[3]=(_Float16)fc.w;
                    p1[4]=(_Float16)fd.x; p1[5]=(_Float16)fd.y;
                    p1[6]=(_Float16)fd.z; p1[7]=(_Float16)fd.w;
                    *(v8h*)(dst + seg * 16)     = p0;
                    *(v8h*)(dst + seg * 16 + 8) = p1;
                }
            }
        }
        __syncthreads();
        // ---- compute: every wave sweeps all staged tiles ----
#pragma unroll 4
        for (int tl = 0; tl < STAGE_TILES; ++tl) {
            if (sb + tl >= t1) break;           // block-uniform
            const _Float16* rb = &buf[(tl * 16 + m16) * ROW_STRIDE];
            // A layout: elem e -> K = 16h + e; a1 adds K+32.
            v8h s0 = *(const v8h*)(rb + 16 * h);
            v8h s1 = *(const v8h*)(rb + 8 + 16 * h);
            v8h s2 = *(const v8h*)(rb + 32 + 16 * h);
            v8h s3 = *(const v8h*)(rb + 40 + 16 * h);
            v16h a0 = __builtin_shufflevector(s0, s1, 0,1,2,3,4,5,6,7,
                                              8,9,10,11,12,13,14,15);
            v16h a1 = __builtin_shufflevector(s2, s3, 0,1,2,3,4,5,6,7,
                                              8,9,10,11,12,13,14,15);

            // Issue all 4 WMMAs back-to-back, then filter: the c0 filter VALU
            // covers c1's WMMA->VALU hazard window (no v_nop padding).
            v8f c0 = {}, c1 = {};
            c0 = __builtin_amdgcn_wmma_f32_16x16x32_f16(false, a0, false, bq[0][0],
                                                        (short)0, c0, false, false);
            c0 = __builtin_amdgcn_wmma_f32_16x16x32_f16(false, a1, false, bq[0][1],
                                                        (short)0, c0, false, false);
            c1 = __builtin_amdgcn_wmma_f32_16x16x32_f16(false, a0, false, bq[1][0],
                                                        (short)0, c1, false, false);
            c1 = __builtin_amdgcn_wmma_f32_16x16x32_f16(false, a1, false, bq[1][1],
                                                        (short)0, c1, false, false);

            const int idbase = (sb + tl) * 16 + 8 * h;
            filter8(c0, idbase, ts[0], ti[0]);
            filter8(c1, idbase, ts[1], ti[1]);
        }
    }

    // Merge lane pairs (l, l+16) sharing a query via ds_swizzle SWAPX16.
#pragma unroll
    for (int g = 0; g < QG; ++g) {
        float ps[TOPK]; int pi[TOPK];
#pragma unroll
        for (int j = 0; j < TOPK; ++j) {
            ps[j] = __int_as_float(__builtin_amdgcn_ds_swizzle(
                        __float_as_int(ts[g][j]), 0x401F));
            pi[j] = __builtin_amdgcn_ds_swizzle(ti[g][j], 0x401F);
        }
        if (h == 0) {
#pragma unroll
            for (int j = 0; j < TOPK; ++j)
                if (ps[j] > ts[g][TOPK - 1]) ins10(ts[g], ti[g], ps[j], pi[j]);
        }
    }

    // Waves own disjoint queries -> write per-slice candidates directly.
    if (h == 0) {
#pragma unroll
        for (int g = 0; g < QG; ++g) {
            const int qg = (wv * QG + g) * 16 + m16;
            size_t base = ((size_t)qg * nslices + slice) * TOPK;
#pragma unroll
            for (int j = 0; j < TOPK; ++j) {
                cand_s[base + j] = ts[g][j];
                cand_i[base + j] = ti[g][j];
            }
        }
    }
}

// ---------------------------------------------------------------------------
// Kernel 3: per-query merge of nslices*10 candidates; gather doc ids.
// Output layout: doc_ids[256*10] (as float) then scores[256*10].
// ---------------------------------------------------------------------------
__global__ __launch_bounds__(32)
void final_topk_kernel(const float* __restrict__ cand_s,
                       const int* __restrict__ cand_i,
                       const long long* __restrict__ doc_ids,
                       float* __restrict__ out, int nslices) {
    const int q = blockIdx.x;
    const int lane = threadIdx.x;
    const int total = nslices * TOPK;

    float s[TOPK]; int ix[TOPK];
#pragma unroll
    for (int j = 0; j < TOPK; ++j) { s[j] = -3.0e38f; ix[j] = 0; }

    const size_t base = (size_t)q * total;
    for (int e = lane; e < total; e += 32) {
        float sc = cand_s[base + e];
        if (sc > s[TOPK - 1]) ins10(s, ix, sc, cand_i[base + e]);
    }

    __shared__ float lsf[32][TOPK];
    __shared__ int   lii[32][TOPK];
#pragma unroll
    for (int j = 0; j < TOPK; ++j) { lsf[lane][j] = s[j]; lii[lane][j] = ix[j]; }
    __syncthreads();

    if (lane == 0) {
        float ms[TOPK]; int mi[TOPK];
#pragma unroll
        for (int j = 0; j < TOPK; ++j) { ms[j] = lsf[0][j]; mi[j] = lii[0][j]; }
        for (int l2 = 1; l2 < 32; ++l2) {
#pragma unroll
            for (int j = 0; j < TOPK; ++j) {
                float sc = lsf[l2][j];
                if (sc > ms[TOPK - 1]) ins10(ms, mi, sc, lii[l2][j]);
            }
        }
#pragma unroll
        for (int j = 0; j < TOPK; ++j) {
            out[(size_t)q * TOPK + j] = (float)doc_ids[mi[j]];
            out[(size_t)NB * TOPK + (size_t)q * TOPK + j] = ms[j];
        }
    }
}

// ---------------------------------------------------------------------------
extern "C" void kernel_launch(void* const* d_in, const int* in_sizes, int n_in,
                              void* d_out, int out_size, void* d_ws, size_t ws_size,
                              hipStream_t stream) {
    (void)in_sizes; (void)n_in; (void)out_size;
    const float*     queries = (const float*)d_in[0];
    const float*     w_proj  = (const float*)d_in[1];
    const float*     corpus  = (const float*)d_in[2];
    const long long* doc_ids = (const long long*)d_in[3];
    float* out = (float*)d_out;

    // Workspace layout: [0,32K) proj f16 ; [64K,..) cand scores ; then cand idx
    _Float16* proj = (_Float16*)d_ws;
    const size_t cand_off = 64 * 1024;
    long avail = (ws_size > cand_off) ? (long)(ws_size - cand_off) : 0;
    int nslices = (int)(avail / (long)(NB * TOPK * 4 * 2));
    if (nslices > MAX_SLICES) nslices = MAX_SLICES;
    if (nslices < 1) nslices = 1;
    float* cand_s = (float*)((char*)d_ws + cand_off);
    int*   cand_i = (int*)((char*)d_ws + cand_off + (size_t)NB * nslices * TOPK * 4);
    const int tps = (TILES_TOTAL + nslices - 1) / nslices;

    proj_kernel<<<1, 256, 0, stream>>>(queries, w_proj, proj);
    score_topk_kernel<<<nslices, 256, 0, stream>>>(corpus, proj, cand_s, cand_i,
                                                   nslices, tps);
    final_topk_kernel<<<NB, 32, 0, stream>>>(cand_s, cand_i, doc_ids, out, nslices);
}
